// VisionMambaEncoder_73796128080033
// MI455X (gfx1250) — matrix operand order repaired and checked by reading, not compiled, not run on hardware
//
#include <hip/hip_runtime.h>
#include <math.h>

#define DEPTH   24
#define EMBED   384
#define D_INNER 768
#define D_STATE 16
#define DT_RANK 24
#define D_CONV  4
#define GH      14
#define GW      14
#define N_PATCH 196
#define SEQ     197
#define BATCH   32
#define EPS     1e-5f
#define XPROJ_N (DT_RANK + 2*D_STATE)   // 56
#define XLD     64                      // padded xdbl leading dim
#define TOK     (BATCH*SEQ)             // 6304
#define TOK_PAD 6400                    // 50 * 128
#define PTOK    (BATCH*N_PATCH)         // 6272 = 49 * 128

typedef __bf16 bf16;
typedef __attribute__((ext_vector_type(4)))  __bf16 v4bf;
typedef __attribute__((ext_vector_type(16))) __bf16 v16bf;
typedef __attribute__((ext_vector_type(8)))  float  v8f;

__device__ __forceinline__ float dev_silu(float x) { return x / (1.f + __expf(-x)); }
__device__ __forceinline__ float dev_softplus(float x) {
  return (x > 20.f) ? x : log1pf(__expf(x));
}

// ---------------------------------------------------------------------------
// bf16 WMMA GEMM:  C[M,N] = A[M,K] * W[N,K]^T (+bias[n]) (+softplus)
// REQUIRES: M % 128 == 0, N % 64 == 0, K % 32 == 0 (caller pads).
//
// Block: 256 threads = 8 waves, tile 128x64, K-step 32, LDS double-buffered.
// Wave (wm 0..3, wn 0..1): 32x32 subtile = 4 x v_wmma_f32_16x16x32_bf16.
//
// LDS tiles pre-swizzled to fragment order [frag][lane][16 bf16]:
//   A (contiguous k order, code's choice): lane = (r&15) + 16*(k>=16), elem = k&15
//   B (same contiguous k order as A): lane = (n&15) + 16*(k>=16), elem = k&15
// A K-aligned float4 lands in 4 consecutive elems of one lane for both ->
// staging = float4 global load + v4bf (ds_store_b64); operands = v16bf
// (2x ds_load_b128). Branch-free inner loop, one barrier per K-step.
// ---------------------------------------------------------------------------
#define BM 128
#define BN 64
#define BK 32
#define AQ ((BM*BK)/(4*256))   // 4 float4 per thread
#define BQ ((BK*BN)/(4*256))   // 2 float4 per thread

__global__ __launch_bounds__(256) void gemm_bf16_wmma(
    const float* __restrict__ A, int lda,
    const float* __restrict__ W, int ldw,      // W stored N x K (row-major)
    float* __restrict__ C, int ldc,
    int M, int N, int K,
    const float* __restrict__ bias, int act)
{
  __shared__ v16bf sAf[2][8 * 32];   // 2 x 8KB
  __shared__ v16bf sBf[2][4 * 32];   // 2 x 4KB

  const int tid  = threadIdx.x;
  const int lane = tid & 31;
  const int wave = tid >> 5;
  const int wm   = wave >> 1;   // 0..3
  const int wn   = wave & 1;    // 0..1
  const long blockRow = (long)blockIdx.y * BM;
  const long blockCol = (long)blockIdx.x * BN;

  v8f acc00 = {}, acc01 = {}, acc10 = {}, acc11 = {};
  float4 va[AQ], vb[BQ];

  auto load_global = [&](int k0) {
    #pragma unroll
    for (int i = 0; i < AQ; ++i) {
      const int q = tid + i * 256;
      const int r = q >> 3, k4 = (q & 7) << 2;
      va[i] = *(const float4*)(A + (blockRow + r) * (long)lda + k0 + k4);
    }
    #pragma unroll
    for (int i = 0; i < BQ; ++i) {
      const int q = tid + i * 256;
      const int n = q >> 3, k4 = (q & 7) << 2;
      vb[i] = *(const float4*)(W + (blockCol + n) * (long)ldw + k0 + k4);
    }
  };
  auto store_lds = [&](int buf) {
    bf16* sA = (bf16*)sAf[buf];
    bf16* sB = (bf16*)sBf[buf];
    #pragma unroll
    for (int i = 0; i < AQ; ++i) {
      const int q = tid + i * 256;
      const int r = q >> 3, k4 = (q & 7) << 2;
      const int fl = (r & 15) + ((k4 & 16) ? 16 : 0);
      const int e0 = (k4 & 15);
      v4bf c;
      c[0] = (bf16)va[i].x; c[1] = (bf16)va[i].y;
      c[2] = (bf16)va[i].z; c[3] = (bf16)va[i].w;
      *(v4bf*)&sA[((((r >> 4) << 5) + fl) << 4) + e0] = c;
    }
    #pragma unroll
    for (int i = 0; i < BQ; ++i) {
      const int q = tid + i * 256;
      const int n = q >> 3, k4 = (q & 7) << 2;
      const int fl = (n & 15) + ((k4 & 16) ? 16 : 0);
      const int e0 = (k4 & 15);
      v4bf c;
      c[0] = (bf16)vb[i].x; c[1] = (bf16)vb[i].y;
      c[2] = (bf16)vb[i].z; c[3] = (bf16)vb[i].w;
      *(v4bf*)&sB[((((n >> 4) << 5) + fl) << 4) + e0] = c;
    }
  };

  load_global(0);
  store_lds(0);
  __syncthreads();

  int buf = 0;
  for (int k0 = 0; k0 < K; k0 += BK) {
    const int nxt = k0 + BK;
    if (nxt < K) load_global(nxt);   // overlap next tile's global loads with WMMA

    const v16bf a0 = sAf[buf][((wm * 2 + 0) << 5) + lane];
    const v16bf a1 = sAf[buf][((wm * 2 + 1) << 5) + lane];
    const v16bf b0 = sBf[buf][((wn * 2 + 0) << 5) + lane];
    const v16bf b1 = sBf[buf][((wn * 2 + 1) << 5) + lane];

    acc00 = __builtin_amdgcn_wmma_f32_16x16x32_bf16(false, a0, false, b0, (short)0, acc00, false, false);
    acc01 = __builtin_amdgcn_wmma_f32_16x16x32_bf16(false, a0, false, b1, (short)0, acc01, false, false);
    acc10 = __builtin_amdgcn_wmma_f32_16x16x32_bf16(false, a1, false, b0, (short)0, acc10, false, false);
    acc11 = __builtin_amdgcn_wmma_f32_16x16x32_bf16(false, a1, false, b1, (short)0, acc11, false, false);

    if (nxt < K) {
      store_lds(buf ^ 1);
      __syncthreads();
      buf ^= 1;
    }
  }

  // epilogue: D layout VGPR r -> M=r (lanes 0-15) / M=8+r (lanes 16-31), N=lane&15
  const int n_l   = lane & 15;
  const int m_off = (lane >> 4) * 8;
  #pragma unroll
  for (int mi = 0; mi < 2; ++mi) {
    #pragma unroll
    for (int ni = 0; ni < 2; ++ni) {
      const v8f acc = mi ? (ni ? acc11 : acc10) : (ni ? acc01 : acc00);
      const long nb = blockCol + wn * 32 + ni * 16 + n_l;
      const long mb = blockRow + wm * 32 + mi * 16 + m_off;
      const float bv = bias ? bias[nb] : 0.f;
      #pragma unroll
      for (int rr = 0; rr < 8; ++rr) {
        float v = acc[rr] + bv;
        if (act == 1) v = dev_softplus(v);
        C[(mb + rr) * (long)ldc + nb] = v;
      }
    }
  }
}

// ---------------------------------------------------------------------------
// Zero-padded weight staging (once per launch, all layers at once)
// ---------------------------------------------------------------------------
__global__ void pad_xproj_w(const float* __restrict__ src, float* __restrict__ dst) {
  long i = (long)blockIdx.x * 256 + threadIdx.x;
  const long total = (long)DEPTH * XLD * D_INNER;
  if (i >= total) return;
  int k = (int)(i % D_INNER);
  long t = i / D_INNER;
  int n = (int)(t % XLD);
  int layer = (int)(t / XLD);
  dst[i] = (n < XPROJ_N) ? src[((long)layer * XPROJ_N + n) * D_INNER + k] : 0.f;
}

__global__ void pad_dtproj_w(const float* __restrict__ src, float* __restrict__ dst) {
  long i = (long)blockIdx.x * 256 + threadIdx.x;
  const long total = (long)DEPTH * D_INNER * 32;
  if (i >= total) return;
  int k = (int)(i % 32);
  long t = i / 32;
  int d = (int)(t % D_INNER);
  int layer = (int)(t / D_INNER);
  dst[i] = (k < DT_RANK) ? src[((long)layer * D_INNER + d) * DT_RANK + k] : 0.f;
}

// ---------------------------------------------------------------------------
// Patch im2col:  Ap[m][k], m=b*196+gi*14+gj, k=c*256+p*16+q
// ---------------------------------------------------------------------------
__global__ void im2col_patch(const float* __restrict__ x, float* __restrict__ Ap) {
  long i = (long)blockIdx.x * 256 + threadIdx.x;
  const long total = (long)PTOK * 768;
  if (i >= total) return;
  int k = (int)(i % 768);
  long m = i / 768;
  int c = k >> 8, p = (k >> 4) & 15, q = k & 15;
  int b = (int)(m / N_PATCH);
  int g = (int)(m % N_PATCH);
  int gi = g / GW, gj = g % GW;
  Ap[i] = x[(((long)b * 3 + c) * 224 + gi * 16 + p) * 224 + gj * 16 + q];
}

// ---------------------------------------------------------------------------
// Tokens: insert CLS at position 98, add pos_embed
// ---------------------------------------------------------------------------
__global__ void assemble_tokens(const float* __restrict__ hp, const float* __restrict__ cls,
                                const float* __restrict__ pos, float* __restrict__ hid) {
  long i = (long)blockIdx.x * 256 + threadIdx.x;
  const long total = (long)TOK * EMBED;
  if (i >= total) return;
  int e = (int)(i % EMBED);
  long t = i / EMBED;
  int l = (int)(t % SEQ);
  int b = (int)(t / SEQ);
  float v;
  if (l == 98) v = cls[e];
  else {
    int lp = (l < 98) ? l : l - 1;
    v = hp[((long)b * N_PATCH + lp) * EMBED + e];
  }
  hid[i] = v + pos[(long)l * EMBED + e];
}

// ---------------------------------------------------------------------------
// residual = (first ? hidden : residual + hidden);  out = LN(residual)*w+b
// ---------------------------------------------------------------------------
__global__ __launch_bounds__(128) void add_norm(
    const float* __restrict__ hidden, float* __restrict__ residual,
    const float* __restrict__ w, const float* __restrict__ b,
    float* __restrict__ out, int first)
{
  const long t = blockIdx.x;
  const float* hrow = hidden + t * EMBED;
  float* rrow = residual + t * EMBED;
  __shared__ float red[2][4];
  float local[3];
  float s = 0.f, s2 = 0.f;
  #pragma unroll
  for (int j = 0; j < 3; ++j) {
    int e = threadIdx.x + j * 128;
    float r = hrow[e] + (first ? 0.f : rrow[e]);
    local[j] = r;
    s += r; s2 += r * r;
  }
  #pragma unroll
  for (int off = 16; off; off >>= 1) {
    s  += __shfl_down(s,  off, 32);
    s2 += __shfl_down(s2, off, 32);
  }
  int lane = threadIdx.x & 31, wv = threadIdx.x >> 5;
  if (lane == 0) { red[0][wv] = s; red[1][wv] = s2; }
  __syncthreads();
  if (threadIdx.x == 0) {
    float ts = 0.f, ts2 = 0.f;
    #pragma unroll
    for (int i = 0; i < 4; ++i) { ts += red[0][i]; ts2 += red[1][i]; }
    red[0][0] = ts; red[1][0] = ts2;
  }
  __syncthreads();
  float mean = red[0][0] * (1.f / EMBED);
  float var  = red[1][0] * (1.f / EMBED) - mean * mean;
  float rinv = rsqrtf(var + EPS);
  #pragma unroll
  for (int j = 0; j < 3; ++j) {
    int e = threadIdx.x + j * 128;
    rrow[e] = local[j];
    out[t * EMBED + e] = (local[j] - mean) * rinv * w[e] + b[e];
  }
}

// ---------------------------------------------------------------------------
// Causal conv1d (width 4) + SiLU, forward + reversed directions in one pass.
// ---------------------------------------------------------------------------
__global__ void conv_silu_both(
    const float* __restrict__ x, int ldx,
    const float* __restrict__ cw,  const float* __restrict__ cb,
    const float* __restrict__ cwb, const float* __restrict__ cbb,
    float* __restrict__ xcf, float* __restrict__ xcb)
{
  long i = (long)blockIdx.x * 256 + threadIdx.x;
  const long total = (long)TOK * D_INNER;
  if (i >= total) return;
  int d = (int)(i % D_INNER);
  long t = i / D_INNER;
  int l = (int)(t % SEQ);
  int b = (int)(t / SEQ);
  const float* xb = x + (long)b * SEQ * ldx;

  float accf = cb[d];
  float accb = cbb[d];
  #pragma unroll
  for (int k = 0; k < D_CONV; ++k) {
    int ls = l - (D_CONV - 1) + k;
    if (ls >= 0) {
      accf += xb[(long)ls * ldx + d] * cw[d * D_CONV + k];
      accb += xb[(long)(SEQ - 1 - ls) * ldx + d] * cwb[d * D_CONV + k];
    }
  }
  xcf[i] = dev_silu(accf);
  xcb[i] = dev_silu(accb);
}

// ---------------------------------------------------------------------------
// Selective scan: one thread per (b,d) channel, 16 states in registers.
// xdbl rows have stride XLD=64; B at cols 24..39, C at cols 40..55.
// ---------------------------------------------------------------------------
__global__ __launch_bounds__(256) void ssm_scan_kernel(
    const float* __restrict__ xc,    // [B,SEQ,768]
    const float* __restrict__ dt,    // [B,SEQ,768] (softplus applied)
    const float* __restrict__ xdbl,  // [B,SEQ,XLD]
    const float* __restrict__ Alog,  // [768,16]
    const float* __restrict__ Dv,    // [768]
    float* __restrict__ y)           // [B,SEQ,768]
{
  const int b = blockIdx.y;
  const int d = blockIdx.x * 256 + threadIdx.x;
  __shared__ float sBC[2 * D_STATE];

  float A[D_STATE], h[D_STATE];
  #pragma unroll
  for (int n = 0; n < D_STATE; ++n) {
    A[n] = -__expf(Alog[(long)d * D_STATE + n]);
    h[n] = 0.f;
  }
  const float Dd = Dv[d];

  const float* xcb = xc   + (long)b * SEQ * D_INNER;
  const float* dtb = dt   + (long)b * SEQ * D_INNER;
  const float* xdb = xdbl + (long)b * SEQ * XLD;
  float* yb = y + (long)b * SEQ * D_INNER;

  for (int l = 0; l < SEQ; ++l) {
    __syncthreads();
    if (threadIdx.x < 2 * D_STATE)
      sBC[threadIdx.x] = xdb[(long)l * XLD + DT_RANK + threadIdx.x];
    __syncthreads();
    const float dtv = dtb[(long)l * D_INNER + d];
    const float u   = xcb[(long)l * D_INNER + d];
    const float du  = dtv * u;
    float yv = 0.f;
    #pragma unroll
    for (int n = 0; n < D_STATE; ++n) {
      h[n] = h[n] * __expf(dtv * A[n]) + du * sBC[n];
      yv += h[n] * sBC[D_STATE + n];
    }
    yb[(long)l * D_INNER + d] = yv + u * Dd;
  }
}

// ---------------------------------------------------------------------------
// y_comb[b,l,d] = 0.5 * silu(z[b,l,d]) * (y_f[b,l,d] + y_b[b,SEQ-1-l,d])
// ---------------------------------------------------------------------------
__global__ void combine_gate(
    const float* __restrict__ yf, const float* __restrict__ ybr,
    const float* __restrict__ z, int ldz, float* __restrict__ out)
{
  long i = (long)blockIdx.x * 256 + threadIdx.x;
  const long total = (long)TOK * D_INNER;
  if (i >= total) return;
  int d = (int)(i % D_INNER);
  long t = i / D_INNER;
  int l = (int)(t % SEQ);
  int b = (int)(t / SEQ);
  float zv = z[((long)b * SEQ + l) * ldz + d];
  float ybv = ybr[((long)b * SEQ + (SEQ - 1 - l)) * D_INNER + d];
  out[i] = 0.5f * dev_silu(zv) * (yf[i] + ybv);
}

// ---------------------------------------------------------------------------
// feat[b,e,gi,gj] = hidden[b, l(g), e]  (skip CLS token)
// ---------------------------------------------------------------------------
__global__ void extract_feat(const float* __restrict__ hidden, float* __restrict__ feat) {
  long i = (long)blockIdx.x * 256 + threadIdx.x;
  const long total = (long)BATCH * EMBED * N_PATCH;
  if (i >= total) return;
  int g = (int)(i % N_PATCH);
  long t = i / N_PATCH;
  int e = (int)(t % EMBED);
  int b = (int)(t / EMBED);
  int l = (g < 98) ? g : g + 1;
  feat[i] = hidden[((long)b * SEQ + l) * EMBED + e];
}

// ---------------------------------------------------------------------------

static inline void launch_gemm(hipStream_t s, const float* A, int lda,
                               const float* W, int ldw, float* C, int ldc,
                               int M, int N, int K, const float* bias, int act) {
  dim3 g(N / BN, M / BM);
  gemm_bf16_wmma<<<g, 256, 0, s>>>(A, lda, W, ldw, C, ldc, M, N, K, bias, act);
}

extern "C" void kernel_launch(void* const* d_in, const int* in_sizes, int n_in,
                              void* d_out, int out_size, void* d_ws, size_t ws_size,
                              hipStream_t stream) {
  const float* x_img     = (const float*)d_in[0];
  const float* patch_w   = (const float*)d_in[1];
  const float* patch_b   = (const float*)d_in[2];
  const float* cls_tok   = (const float*)d_in[3];
  const float* pos_embed = (const float*)d_in[4];
  const float* norm_w    = (const float*)d_in[5];
  const float* norm_b    = (const float*)d_in[6];
  const float* in_proj_w = (const float*)d_in[7];
  const float* conv_w    = (const float*)d_in[8];
  const float* conv_b    = (const float*)d_in[9];
  const float* xproj_w   = (const float*)d_in[10];
  const float* dtproj_w  = (const float*)d_in[11];
  const float* dtproj_b  = (const float*)d_in[12];
  const float* A_log     = (const float*)d_in[13];
  const float* D_param   = (const float*)d_in[14];
  const float* conv_w_b  = (const float*)d_in[15];
  const float* conv_b_b  = (const float*)d_in[16];
  const float* xproj_w_b = (const float*)d_in[17];
  const float* dtproj_w_b= (const float*)d_in[18];
  const float* dtproj_b_b= (const float*)d_in[19];
  const float* A_log_b   = (const float*)d_in[20];
  const float* D_param_b = (const float*)d_in[21];
  const float* out_proj_w= (const float*)d_in[22];
  const float* normf_w   = (const float*)d_in[23];
  const float* normf_b   = (const float*)d_in[24];

  float* out = (float*)d_out;

  float* ws = (float*)d_ws;
  size_t off = 0;
  auto alloc = [&](size_t n) { float* p = ws + off; off += n; return p; };

  float* Ap    = alloc((size_t)PTOK * 768);
  float* hp    = alloc((size_t)PTOK * EMBED);
  float* hidden= alloc((size_t)TOK_PAD * EMBED);
  float* resid = alloc((size_t)TOK_PAD * EMBED);
  float* hn    = alloc((size_t)TOK_PAD * EMBED);
  float* xz    = alloc((size_t)TOK_PAD * 2 * D_INNER);
  float* xcf   = alloc((size_t)TOK_PAD * D_INNER);
  float* xcb   = alloc((size_t)TOK_PAD * D_INNER);
  float* xdf   = alloc((size_t)TOK_PAD * XLD);
  float* xdb   = alloc((size_t)TOK_PAD * XLD);
  float* dtf   = alloc((size_t)TOK_PAD * D_INNER);
  float* dtb   = alloc((size_t)TOK_PAD * D_INNER);
  float* yfb   = alloc((size_t)TOK_PAD * D_INNER);
  float* ybb   = alloc((size_t)TOK_PAD * D_INNER);
  float* ycomb = alloc((size_t)TOK_PAD * D_INNER);
  float* xwpf  = alloc((size_t)DEPTH * XLD * D_INNER);   // padded x_proj weights
  float* xwpb  = alloc((size_t)DEPTH * XLD * D_INNER);
  float* dtwpf = alloc((size_t)DEPTH * D_INNER * 32);    // padded dt_proj weights
  float* dtwpb = alloc((size_t)DEPTH * D_INNER * 32);
  (void)ws_size;

  const long nImc  = (long)PTOK * 768;
  const long nTokE = (long)TOK * EMBED;
  const long nTokD = (long)TOK * D_INNER;
  const long nFeat = (long)BATCH * EMBED * N_PATCH;
  const long nXW   = (long)DEPTH * XLD * D_INNER;
  const long nDTW  = (long)DEPTH * D_INNER * 32;

  // ---- zero-padded weight staging (all layers) ----
  pad_xproj_w <<<dim3((nXW  + 255) / 256), 256, 0, stream>>>(xproj_w,    xwpf);
  pad_xproj_w <<<dim3((nXW  + 255) / 256), 256, 0, stream>>>(xproj_w_b,  xwpb);
  pad_dtproj_w<<<dim3((nDTW + 255) / 256), 256, 0, stream>>>(dtproj_w,   dtwpf);
  pad_dtproj_w<<<dim3((nDTW + 255) / 256), 256, 0, stream>>>(dtproj_w_b, dtwpb);

  // ---- patch embedding ----
  im2col_patch<<<dim3((nImc + 255) / 256), 256, 0, stream>>>(x_img, Ap);
  launch_gemm(stream, Ap, 768, patch_w, 768, hp, EMBED, PTOK, EMBED, 768, patch_b, 0);
  assemble_tokens<<<dim3((nTokE + 255) / 256), 256, 0, stream>>>(hp, cls_tok, pos_embed, hidden);

  const size_t OUT_TOK = (size_t)TOK * EMBED;
  const size_t FEAT_SZ = (size_t)BATCH * EMBED * N_PATCH;
  int feat_slot = 0;

  for (int i = 0; i < DEPTH; ++i) {
    add_norm<<<dim3(TOK), 128, 0, stream>>>(hidden, resid,
        norm_w + (size_t)i * EMBED, norm_b + (size_t)i * EMBED, hn, (i == 0) ? 1 : 0);

    // in_proj: (6400x384)x(384x1536)
    launch_gemm(stream, hn, EMBED, in_proj_w + (size_t)i * 2 * D_INNER * EMBED, EMBED,
                xz, 2 * D_INNER, TOK_PAD, 2 * D_INNER, EMBED, nullptr, 0);

    conv_silu_both<<<dim3((nTokD + 255) / 256), 256, 0, stream>>>(
        xz, 2 * D_INNER,
        conv_w + (size_t)i * D_INNER * D_CONV,   conv_b + (size_t)i * D_INNER,
        conv_w_b + (size_t)i * D_INNER * D_CONV, conv_b_b + (size_t)i * D_INNER,
        xcf, xcb);

    // x_proj: (6400x768)x(768x64pad)
    launch_gemm(stream, xcf, D_INNER, xwpf + (size_t)i * XLD * D_INNER, D_INNER,
                xdf, XLD, TOK_PAD, XLD, D_INNER, nullptr, 0);
    launch_gemm(stream, xcb, D_INNER, xwpb + (size_t)i * XLD * D_INNER, D_INNER,
                xdb, XLD, TOK_PAD, XLD, D_INNER, nullptr, 0);

    // dt_proj + softplus: (6400x32pad)x(32padx768)
    launch_gemm(stream, xdf, XLD, dtwpf + (size_t)i * D_INNER * 32, 32,
                dtf, D_INNER, TOK_PAD, D_INNER, 32, dtproj_b + (size_t)i * D_INNER, 1);
    launch_gemm(stream, xdb, XLD, dtwpb + (size_t)i * D_INNER * 32, 32,
                dtb, D_INNER, TOK_PAD, D_INNER, 32, dtproj_b_b + (size_t)i * D_INNER, 1);

    // selective scans
    dim3 sg(D_INNER / 256, BATCH);
    ssm_scan_kernel<<<sg, 256, 0, stream>>>(xcf, dtf, xdf,
        A_log + (size_t)i * D_INNER * D_STATE, D_param + (size_t)i * D_INNER, yfb);
    ssm_scan_kernel<<<sg, 256, 0, stream>>>(xcb, dtb, xdb,
        A_log_b + (size_t)i * D_INNER * D_STATE, D_param_b + (size_t)i * D_INNER, ybb);

    combine_gate<<<dim3((nTokD + 255) / 256), 256, 0, stream>>>(
        yfb, ybb, xz + D_INNER, 2 * D_INNER, ycomb);

    // out_proj: (6400x768)x(768x384) -> hidden
    launch_gemm(stream, ycomb, D_INNER, out_proj_w + (size_t)i * EMBED * D_INNER, D_INNER,
                hidden, EMBED, TOK_PAD, EMBED, D_INNER, nullptr, 0);

    if (i == 5 || i == 11 || i == 17 || i == 23) {
      extract_feat<<<dim3((nFeat + 255) / 256), 256, 0, stream>>>(
          hidden, out + OUT_TOK + (size_t)feat_slot * FEAT_SZ);
      feat_slot++;
    }
  }

  add_norm<<<dim3(TOK), 128, 0, stream>>>(hidden, resid, normf_w, normf_b, out, 0);
}
